// MultiheadSelfAttention_72404558676790
// MI455X (gfx1250) — matrix operand order repaired and checked by reading, not compiled, not run on hardware
//
#include <hip/hip_runtime.h>

// ---------------------------------------------------------------------------
// MHA for MI455X (gfx1250): bf16 WMMA pipeline, fp32 accumulate.
//   B=2, S=2048, D=1024, H=16, DH=64
// Data movement: GLOBAL_LOAD_ASYNC_TO_LDS_B128 (ASYNCcnt) + double buffering.
// ---------------------------------------------------------------------------

#define BATCH 2
#define SEQ   2048
#define DMODEL 1024
#define NHEADS 16
#define DHEAD  64

typedef __attribute__((ext_vector_type(16))) __bf16 v16bf;
typedef __attribute__((ext_vector_type(8)))  float  v8f;

union Frag {
    v16bf v;
    uint4 u[2];
    unsigned short s[16];
};
union Frag2 {
    uint4 u[4];
    unsigned short s[32];
};

__device__ __forceinline__ unsigned short f32_to_bf16(float f) {
    unsigned int u = __float_as_uint(f);
    u += 0x7FFFu + ((u >> 16) & 1u);   // round-to-nearest-even
    return (unsigned short)(u >> 16);
}

__device__ __forceinline__ v8f wmma_bf16(v16bf a, v16bf b, v8f c) {
    return __builtin_amdgcn_wmma_f32_16x16x32_bf16(false, a, false, b, (short)0, c, false, false);
}

// ---- gfx1250 async global->LDS copy (ASYNCcnt), 16B per lane ----
__device__ __forceinline__ void async_ld_b128(void* lds_ptr, const void* gptr) {
    const unsigned lds_addr = (unsigned)(size_t)lds_ptr;     // low 32b = LDS offset
    const unsigned long long ga = (unsigned long long)gptr;
    asm volatile("global_load_async_to_lds_b128 %0, %1, off"
                 :: "v"(lds_addr), "v"(ga)
                 : "memory");
}
__device__ __forceinline__ void async_wait0() {
    asm volatile("s_wait_asynccnt 0x0" ::: "memory");
}

// ---------------------------------------------------------------------------
// Kernel 1: fp32 -> bf16 elementwise convert
// ---------------------------------------------------------------------------
__global__ __launch_bounds__(256)
void k_convert_bf16(const float* __restrict__ src, unsigned short* __restrict__ dst, int n) {
    for (int i = blockIdx.x * blockDim.x + threadIdx.x; i < n; i += gridDim.x * blockDim.x)
        dst[i] = f32_to_bf16(src[i]);
}

// ---------------------------------------------------------------------------
// Kernel 2: transpose + convert weight: Wt[k][n] = bf16(W[n][k]),  D x D
// ---------------------------------------------------------------------------
__global__ __launch_bounds__(256)
void k_transpose_bf16(const float* __restrict__ W, unsigned short* __restrict__ Wt, int Dm) {
    __shared__ unsigned short tile[32][33];
    const int tx = threadIdx.x, ty = threadIdx.y;
    const int n0 = blockIdx.y * 32, k0 = blockIdx.x * 32;
#pragma unroll
    for (int i = 0; i < 4; ++i)
        tile[ty + i * 8][tx] = f32_to_bf16(W[(size_t)(n0 + ty + i * 8) * Dm + (k0 + tx)]);
    __syncthreads();
#pragma unroll
    for (int i = 0; i < 4; ++i)
        Wt[(size_t)(k0 + ty + i * 8) * Dm + (n0 + tx)] = tile[tx][ty + i * 8];
}

// ---------------------------------------------------------------------------
// Kernel 3: GEMM  C[M,N] = A[M,K](bf16,row) * Bt[K,N](bf16,row)
//   block = 256 threads = 8 waves; block tile 128M x 128N; wave tile 32x64.
//   Double-buffered LDS tiles filled by async global->LDS copies.
//   8 WMMA per K-step per wave.
//   OUT_F32: 1 -> store fp32, 0 -> store bf16
// ---------------------------------------------------------------------------
template <int OUT_F32>
__global__ __launch_bounds__(256)
void k_gemm_bf16(const unsigned short* __restrict__ A,
                 const unsigned short* __restrict__ Bt,
                 void* __restrict__ Cout, int M, int N, int K) {
    __shared__ __align__(16) unsigned short sA[2][128][40];   // [buf][m][k]
    __shared__ __align__(16) unsigned short sB[2][32][136];   // [buf][k][n]

    const int tid  = threadIdx.x;
    const int lane = tid & 31;
    const int wave = tid >> 5;
    const int wm = wave >> 1, wn = wave & 1;
    const int m0b = blockIdx.y * 128;
    const int n0b = blockIdx.x * 128;
    const int hi = lane >> 4;
    const int nc = lane & 15;

    v8f acc[2][4];
#pragma unroll
    for (int i = 0; i < 2; ++i)
#pragma unroll
        for (int j = 0; j < 4; ++j) acc[i][j] = (v8f)(0.f);

    // staging indices: A tile 128x32 (2x16B / thread), B tile 32x128 (2x16B / thread)
    const int aRow0 = tid >> 2, aQ0 = (tid & 3) * 8;
    const int bRow0 = tid >> 4,            bC0 = (tid & 15) * 8;          // rows 0..15
    const int bRow1 = (tid + 256) >> 4,    bC1 = (tid & 15) * 8;          // rows 16..31

    const unsigned short* aP0 = A + (size_t)(m0b + aRow0) * K + aQ0;
    const unsigned short* aP1 = A + (size_t)(m0b + aRow0 + 64) * K + aQ0;
    const unsigned short* bP0 = Bt + (size_t)bRow0 * N + n0b + bC0;
    const unsigned short* bP1 = Bt + (size_t)bRow1 * N + n0b + bC1;

    auto stage = [&](int buf, int k) {
        async_ld_b128(&sA[buf][aRow0][aQ0],      aP0 + k);
        async_ld_b128(&sA[buf][aRow0 + 64][aQ0], aP1 + k);
        async_ld_b128(&sB[buf][bRow0][bC0],      bP0 + (size_t)k * N);
        async_ld_b128(&sB[buf][bRow1][bC1],      bP1 + (size_t)k * N);
    };

    const int NT = K / 32;
    stage(0, 0);
    async_wait0();
    __syncthreads();

    for (int kt = 0; kt < NT; ++kt) {
        const int buf = kt & 1;
        if (kt + 1 < NT) stage(buf ^ 1, (kt + 1) * 32);       // async next tile
        if (kt + 2 < NT) {                                    // warm L2 two ahead
            __builtin_prefetch(aP0 + (kt + 2) * 32, 0, 1);
            __builtin_prefetch(bP0 + (size_t)(kt + 2) * 32 * N, 0, 1);
        }

        // ---- fragments from LDS (distinct regs) + 8 WMMA ----
        v16bf a[2], b[4];
#pragma unroll
        for (int i = 0; i < 2; ++i) {
            Frag f;
            const unsigned short* p = &sA[buf][wm * 32 + i * 16 + nc][hi << 3];
            f.u[0] = *reinterpret_cast<const uint4*>(p);
            f.u[1] = *reinterpret_cast<const uint4*>(p + 16);
            a[i] = f.v;
        }
#pragma unroll
        for (int j = 0; j < 4; ++j) {
            Frag f;
#pragma unroll
            for (int s = 0; s < 16; ++s)
                f.s[s] = sB[buf][(hi << 3) + (s & 7) + ((s >> 3) << 4)][wn * 64 + j * 16 + nc];
            b[j] = f.v;
        }
#pragma unroll
        for (int i = 0; i < 2; ++i)
#pragma unroll
            for (int j = 0; j < 4; ++j)
                acc[i][j] = wmma_bf16(a[i], b[j], acc[i][j]);

        async_wait0();       // next tile resident in LDS
        __syncthreads();     // all waves done reading buf before reuse
    }

    // C layout: lanes 0-15 -> N=lane, M=r; lanes 16-31 -> N=lane-16, M=r+8
    const int m0 = m0b + wm * 32;
    const int n0 = n0b + wn * 64;
#pragma unroll
    for (int i = 0; i < 2; ++i)
#pragma unroll
        for (int j = 0; j < 4; ++j)
#pragma unroll
            for (int r = 0; r < 8; ++r) {
                const int row = m0 + i * 16 + hi * 8 + r;
                const int col = n0 + j * 16 + nc;
                const float val = acc[i][j][r];
                if (OUT_F32)
                    reinterpret_cast<float*>(Cout)[(size_t)row * N + col] = val;
                else
                    reinterpret_cast<unsigned short*>(Cout)[(size_t)row * N + col] = f32_to_bf16(val);
            }
}

// ---------------------------------------------------------------------------
// Kernel 4: causal flash attention, transposed-scores formulation.
//   S^T = K(16k x 32dh) x Q^T(32dh x 16q)  -> C layout of S^T IS the
//   A-fragment layout of P, so PV consumes P directly from registers.
//   Double-buffered K/V tiles staged with async global->LDS copies,
//   overlapped with softmax VALU work.
//   grid = (S/64, H, B), block = 128 threads (4 waves, 16 q-rows each)
// ---------------------------------------------------------------------------
__global__ __launch_bounds__(128)
void k_attention(const unsigned short* __restrict__ Q,
                 const unsigned short* __restrict__ Kg,
                 const unsigned short* __restrict__ V,
                 unsigned short* __restrict__ Ctx) {
    __shared__ __align__(16) unsigned short sQt[64][72];     // Q^T [dh][q]
    __shared__ __align__(16) unsigned short sKr[2][32][72];  // K tile [buf][k][dh]
    __shared__ __align__(16) unsigned short sVr[2][32][72];  // V tile [buf][k][dh]

    const int qBase = blockIdx.x * 64, h = blockIdx.y, b = blockIdx.z;
    const int tid  = threadIdx.x;
    const int wave = tid >> 5;
    const int lane = tid & 31;
    const int hi = lane >> 4;
    const int nc = lane & 15;
    const int qw = qBase + wave * 16;
    const size_t headOff = (size_t)h * DHEAD;

    // ---- stage Q^T once: sQt[d][q] = Q[qBase+q][d] ----
    {
        const int qrow = tid & 63;
        const int chunk = (tid >> 6) * 32;
        const unsigned short* p = Q + ((size_t)b * SEQ + qBase + qrow) * DMODEL + headOff + chunk;
        Frag2 f;
        f.u[0] = *reinterpret_cast<const uint4*>(p);
        f.u[1] = *reinterpret_cast<const uint4*>(p + 8);
        f.u[2] = *reinterpret_cast<const uint4*>(p + 16);
        f.u[3] = *reinterpret_cast<const uint4*>(p + 24);
#pragma unroll
        for (int i = 0; i < 32; ++i) sQt[chunk + i][qrow] = f.s[i];
    }
    __syncthreads();

    // ---- per-wave Q^T B-fragments (loop invariant): lane = q col, 16 d elems ----
    v16bf bQ[2];
#pragma unroll
    for (int c = 0; c < 2; ++c) {
        Frag f;
#pragma unroll
        for (int s = 0; s < 16; ++s)
            f.s[s] = sQt[c * 32 + (hi << 3) + (s & 7) + ((s >> 3) << 4)][wave * 16 + nc];
        bQ[c] = f.v;
    }

    // per-lane softmax state for q = qw + nc
    const int q_abs = qw + nc;
    float rowmax = -3.0e38f, rowsum = 0.f;
    v8f accO[4];
#pragma unroll
    for (int t = 0; t < 4; ++t) accO[t] = (v8f)(0.f);
    const float scale = 0.125f;  // 1/sqrt(64)

    const int kvRow = tid >> 2;
    const int kvC   = (tid & 3) * 16;
    const unsigned short* kSrc = Kg + ((size_t)b * SEQ + kvRow) * DMODEL + headOff + kvC;
    const unsigned short* vSrc = V + ((size_t)b * SEQ + kvRow) * DMODEL + headOff + kvC;

    auto stageKV = [&](int buf, int kb) {
        const size_t off = (size_t)kb * DMODEL;
        async_ld_b128(&sKr[buf][kvRow][kvC],     kSrc + off);
        async_ld_b128(&sKr[buf][kvRow][kvC + 8], kSrc + off + 8);
        async_ld_b128(&sVr[buf][kvRow][kvC],     vSrc + off);
        async_ld_b128(&sVr[buf][kvRow][kvC + 8], vSrc + off + 8);
    };

    const int kEnd = qBase + 64;
    stageKV(0, 0);
    async_wait0();
    __syncthreads();

    for (int kb = 0; kb < kEnd; kb += 32) {
        const int buf = (kb >> 5) & 1;
        if (kb + 32 < kEnd) stageKV(buf ^ 1, kb + 32);   // async next tile

        // ---- scores^T: preload 4 A-frags, then 4 WMMA ----
        Frag aK[2][2];
#pragma unroll
        for (int j = 0; j < 2; ++j)
#pragma unroll
            for (int c = 0; c < 2; ++c) {
                const unsigned short* p = &sKr[buf][j * 16 + nc][c * 32 + (hi << 3)];
                aK[j][c].u[0] = *reinterpret_cast<const uint4*>(p);
                aK[j][c].u[1] = *reinterpret_cast<const uint4*>(p + 16);
            }
        v8f sT[2];
        sT[0] = (v8f)(0.f);
        sT[1] = (v8f)(0.f);
#pragma unroll
        for (int j = 0; j < 2; ++j)
#pragma unroll
            for (int c = 0; c < 2; ++c)
                sT[j] = wmma_bf16(aK[j][c].v, bQ[c], sT[j]);

        // ---- per-lane online softmax over 16 key values (this lane's q) ----
        float sv[2][8];
        float tm = -3.0e38f;
#pragma unroll
        for (int j = 0; j < 2; ++j)
#pragma unroll
            for (int r = 0; r < 8; ++r) {
                const int key = kb + 16 * j + r + 8 * hi;
                float s = sT[j][r] * scale;
                s = (key <= q_abs) ? s : -3.0e38f;
                sv[j][r] = s;
                tm = fmaxf(tm, s);
            }
        tm = fmaxf(tm, __shfl_xor(tm, 16));
        const float nm = fmaxf(rowmax, tm);
        const float alpha = __expf(rowmax - nm);

        Frag pf;  // P in A-fragment layout (== S^T C layout), bf16
        float ps = 0.f;
#pragma unroll
        for (int r = 0; r < 8; ++r) {
            const bool ok0 = (kb + r + 8 * hi)      <= q_abs;
            const bool ok1 = (kb + 16 + r + 8 * hi) <= q_abs;
            const float e0 = ok0 ? __expf(sv[0][r] - nm) : 0.f;
            const float e1 = ok1 ? __expf(sv[1][r] - nm) : 0.f;
            ps += e0 + e1;
            pf.s[r]     = f32_to_bf16(e0);
            pf.s[r + 8] = f32_to_bf16(e1);
        }
        ps += __shfl_xor(ps, 16);
        rowsum = rowsum * alpha + ps;
        rowmax = nm;

        // ---- rescale ctx accumulators (alpha indexed by q = r + 8*hi) ----
#pragma unroll
        for (int r = 0; r < 8; ++r) {
            const float av = __shfl(alpha, r + 8 * hi);
#pragma unroll
            for (int t = 0; t < 4; ++t) accO[t][r] *= av;
        }

        // ---- PV: preload 4 V B-frags, then 4 WMMA ----
        Frag bV[4];
#pragma unroll
        for (int t = 0; t < 4; ++t) {
#pragma unroll
            for (int s = 0; s < 16; ++s)
                bV[t].s[s] = sVr[buf][(hi << 3) + (s & 7) + ((s >> 3) << 4)][t * 16 + nc];
        }
#pragma unroll
        for (int t = 0; t < 4; ++t)
            accO[t] = wmma_bf16(pf.v, bV[t].v, accO[t]);

        async_wait0();       // next K/V tile resident
        __syncthreads();     // all waves done reading buf before reuse
    }

    // ---- finalize: divide by rowsum (broadcast per q), store ctx bf16 ----
    float sumr[8];
#pragma unroll
    for (int r = 0; r < 8; ++r) sumr[r] = __shfl(rowsum, r + 8 * hi);
#pragma unroll
    for (int t = 0; t < 4; ++t)
#pragma unroll
        for (int r = 0; r < 8; ++r) {
            const float val = accO[t][r] / sumr[r];
            Ctx[((size_t)b * SEQ + qw + r + 8 * hi) * DMODEL + headOff + t * 16 + nc] =
                f32_to_bf16(val);
        }
}

// ---------------------------------------------------------------------------
// Host-side launch
// ---------------------------------------------------------------------------
extern "C" void kernel_launch(void* const* d_in, const int* in_sizes, int n_in,
                              void* d_out, int out_size, void* d_ws, size_t ws_size,
                              hipStream_t stream) {
    const float* X  = (const float*)d_in[0];
    const float* Wq = (const float*)d_in[1];
    const float* Wk = (const float*)d_in[2];
    const float* Wv = (const float*)d_in[3];
    const float* Wo = (const float*)d_in[4];
    float* out = (float*)d_out;

    char* ws = (char*)d_ws;
    const size_t MB = 1ull << 20;
    unsigned short* Xb  = (unsigned short*)(ws + 0 * MB);   // 8 MB
    unsigned short* WqT = (unsigned short*)(ws + 8 * MB);   // 2 MB each
    unsigned short* WkT = (unsigned short*)(ws + 10 * MB);
    unsigned short* WvT = (unsigned short*)(ws + 12 * MB);
    unsigned short* WoT = (unsigned short*)(ws + 14 * MB);
    unsigned short* Qb  = (unsigned short*)(ws + 16 * MB);  // 8 MB each
    unsigned short* Kb  = (unsigned short*)(ws + 24 * MB);
    unsigned short* Vb  = (unsigned short*)(ws + 32 * MB);
    unsigned short* Ctx = (unsigned short*)(ws + 40 * MB);

    const int nX = BATCH * SEQ * DMODEL;
    k_convert_bf16<<<2048, 256, 0, stream>>>(X, Xb, nX);

    dim3 tb(32, 8), tg(DMODEL / 32, DMODEL / 32);
    k_transpose_bf16<<<tg, tb, 0, stream>>>(Wq, WqT, DMODEL);
    k_transpose_bf16<<<tg, tb, 0, stream>>>(Wk, WkT, DMODEL);
    k_transpose_bf16<<<tg, tb, 0, stream>>>(Wv, WvT, DMODEL);
    k_transpose_bf16<<<tg, tb, 0, stream>>>(Wo, WoT, DMODEL);

    const int M = BATCH * SEQ;  // 4096
    dim3 gg(DMODEL / 128, M / 128);
    k_gemm_bf16<0><<<gg, 256, 0, stream>>>(Xb, WqT, Qb, M, DMODEL, DMODEL);
    k_gemm_bf16<0><<<gg, 256, 0, stream>>>(Xb, WkT, Kb, M, DMODEL, DMODEL);
    k_gemm_bf16<0><<<gg, 256, 0, stream>>>(Xb, WvT, Vb, M, DMODEL, DMODEL);

    k_attention<<<dim3(SEQ / 64, NHEADS, BATCH), 128, 0, stream>>>(Qb, Kb, Vb, Ctx);

    k_gemm_bf16<1><<<gg, 256, 0, stream>>>(Ctx, WoT, out, M, DMODEL, DMODEL);
}
